// TransformerEncoder_17025250361658
// MI455X (gfx1250) — compile-verified, matrix operand order checked by reading, not run on hardware
//
#include <hip/hip_runtime.h>

// ---------------------------------------------------------------------------
// Types for CDNA5 WMMA (gfx1250, wave32)
// ---------------------------------------------------------------------------
typedef __attribute__((ext_vector_type(16))) _Float16 v16h;
typedef __attribute__((ext_vector_type(8)))  _Float16 v8h;
typedef __attribute__((ext_vector_type(8)))  float    v8f;
typedef __attribute__((ext_vector_type(4)))  unsigned int u32x4;
typedef __attribute__((ext_vector_type(8)))  int      i32x8;
typedef __attribute__((ext_vector_type(4)))  int      i32x4;

#define EMBED   768
#define HEADS   12
#define DEPTH   64
#define HIDDEN  3072
#define SEQ     2048
#define BATCH   4
#define NTOK    (BATCH * SEQ)       // 8192

// Tensor Data Mover availability. This toolchain has the 6-arg builtin
// (probe: "too few arguments to function call, expected 6, have 5").
#if defined(__gfx1250__) && __has_builtin(__builtin_amdgcn_tensor_load_to_lds)
#define USE_TDM 1
#else
#define USE_TDM 0
#endif

__device__ __forceinline__ v16h cat8(v8h lo, v8h hi) {
    return __builtin_shufflevector(lo, hi, 0,1,2,3,4,5,6,7,8,9,10,11,12,13,14,15);
}

__device__ __forceinline__ v16h frag_from_lds(const _Float16* p) {
    // A/B fragment for v_wmma_*_16x16x32: lane holds 8 contiguous halfs at p
    // (K = kb..kb+7) and 8 more at p+16 (K = kb+16..kb+23).
    v8h lo = *(const v8h*)p;
    v8h hi = *(const v8h*)(p + 16);
    return cat8(lo, hi);
}

__device__ __forceinline__ void wait_tensorcnt0() {
#if defined(__gfx1250__)
#if __has_builtin(__builtin_amdgcn_s_wait_tensorcnt)
    __builtin_amdgcn_s_wait_tensorcnt(0);
#else
    asm volatile("s_wait_tensorcnt 0x0" ::: "memory");
#endif
#endif
}

// ---------------------------------------------------------------------------
// f32 -> f16 elementwise (weight conversion)
// ---------------------------------------------------------------------------
__global__ void __launch_bounds__(256)
f32_to_f16_kernel(const float* __restrict__ in, _Float16* __restrict__ out, int n) {
    int i = blockIdx.x * 256 + threadIdx.x;
    if (i < n) out[i] = (_Float16)in[i];
}

// ---------------------------------------------------------------------------
// LayerNorm over 768 (one token per block, optional residual input)
// out (f16) = (x[+res] - mu) * rsqrt(var + eps) * g + b
// ---------------------------------------------------------------------------
__global__ void __launch_bounds__(256)
ln_kernel(const float* __restrict__ x, const float* __restrict__ res,
          const float* __restrict__ g, const float* __restrict__ b,
          _Float16* __restrict__ out) {
    const int row = blockIdx.x, tid = threadIdx.x;
    __shared__ float red[256];
    const size_t base = (size_t)row * EMBED;
    float v[3];
#pragma unroll
    for (int i = 0; i < 3; ++i) {
        int c = tid + i * 256;
        float t = x[base + c];
        if (res) t += res[base + c];
        v[i] = t;
    }
    red[tid] = v[0] + v[1] + v[2];
    __syncthreads();
    for (int off = 128; off > 0; off >>= 1) {
        if (tid < off) red[tid] += red[tid + off];
        __syncthreads();
    }
    float mu = red[0] * (1.0f / EMBED);
    __syncthreads();
    float d0 = v[0] - mu, d1 = v[1] - mu, d2 = v[2] - mu;
    red[tid] = d0 * d0 + d1 * d1 + d2 * d2;
    __syncthreads();
    for (int off = 128; off > 0; off >>= 1) {
        if (tid < off) red[tid] += red[tid + off];
        __syncthreads();
    }
    float rs = rsqrtf(red[0] * (1.0f / EMBED) + 1e-5f);
#pragma unroll
    for (int i = 0; i < 3; ++i) {
        int c = tid + i * 256;
        out[base + c] = (_Float16)((v[i] - mu) * rs * g[c] + b[c]);
    }
}

// ---------------------------------------------------------------------------
// Tiled WMMA GEMM:  C[M,N] = A[M,K] (f16) * B[K,N] (f16) + bias
//   ACT:    0 = none, 1 = exact GELU
//   OUTF32: 0 = f16 output, 1 = f32 output
//   ADDRES: 1 = add f32 residual res[M,N]
// Block = 256 threads (8 waves), tile 128x128, K-step 32.
// Wave grid 2(M) x 4(N): each wave -> 64x32 = 4x2 wmma accumulators.
// A tile is staged by the Tensor Data Mover (wave 0 issues TENSOR_LOAD_TO_LDS
// with LDS padding 16B per 64B row to match As[128][40]); B tile is staged
// transposed by all threads (TDM cannot transpose).
// ---------------------------------------------------------------------------
template<int ACT, int OUTF32, int ADDRES>
__global__ void __launch_bounds__(256)
gemm_wmma(const _Float16* __restrict__ A, const _Float16* __restrict__ Bw,
          const float* __restrict__ bias, const float* __restrict__ res,
          void* __restrict__ Cout, int M, int N, int K) {
    __shared__ _Float16 As[128][40];   // [m][k], padded (+16B per 64B row)
    __shared__ _Float16 Bs[128][40];   // [n][k] (transposed), padded
    const int tid  = threadIdx.x;
    const int lane = tid & 31, wave = tid >> 5;
    const int wm = wave >> 2, wn = wave & 3;      // 2 x 4 waves
    const int m0 = blockIdx.y * 128, n0 = blockIdx.x * 128;
    const int r16 = lane & 15, kb = (lane >> 4) * 8;

    v8f c[4][2];
    {
        v8f z = {};
#pragma unroll
        for (int fm = 0; fm < 4; ++fm)
#pragma unroll
            for (int fn = 0; fn < 2; ++fn) c[fm][fn] = z;
    }

#if USE_TDM
    // Static parts of the TDM descriptor (D#) for the A tile (block-uniform).
    const unsigned int ldsA = (unsigned int)(uintptr_t)&As[0][0];
#endif

    for (int k0 = 0; k0 < K; k0 += 32) {
#if USE_TDM
        // ---- stage A tile 128x32 via Tensor Data Mover (wave 0 only) ----
        if (tid < 32) {
            unsigned long long ga =
                (unsigned long long)(uintptr_t)(A + (size_t)m0 * K + k0);
            u32x4 g0;
            g0[0] = 1u;                                   // count=1, user mode
            g0[1] = ldsA;                                 // lds_addr (bytes)
            g0[2] = (unsigned int)(ga & 0xFFFFFFFFull);   // global_addr[31:0]
            g0[3] = (unsigned int)((ga >> 32) & 0x1FFFFFFull) | (2u << 30); // +type=2

            i32x8 g1;
            g1[0] = (1 << 16)        // data_size = 2 bytes
                  | (1 << 20)        // pad_enable
                  | (3 << 22)        // pad_interval: every 64B
                  | (3 << 25);       // pad_amount: 4 DWORDs = 16B
            g1[1] = (int)((unsigned)K << 16);             // tensor_dim0[15:0]
            g1[2] = (int)(((unsigned)K >> 16) & 0xFFFFu)  // tensor_dim0[31:16]
                  | (int)((unsigned)M << 16);             // tensor_dim1[15:0]
            g1[3] = (int)(((unsigned)M >> 16) & 0xFFFFu)  // tensor_dim1[31:16]
                  | (32 << 16);                           // tile_dim0 = 32
            g1[4] = 128;                                  // tile_dim1 = 128, tile_dim2 = 0
            g1[5] = (int)(unsigned)K;                     // dim0_stride[31:0]
            g1[6] = 0;                                    // dim0_stride[47:32], dim1_stride lo
            g1[7] = 0;

            i32x4 gz4 = {};                               // groups 2/3 unused (2D tensor)
            i32x8 gz8 = {};
            __builtin_amdgcn_tensor_load_to_lds(g0, g1, gz4, gz4, gz8, 0);
            wait_tensorcnt0();
        }
#else
        // ---- stage A tile 128x32 (row-major, b128 copies) ----
#pragma unroll
        for (int i = 0; i < 2; ++i) {
            int idx = tid * 2 + i;            // 0..511
            int row = idx >> 2, c4 = (idx & 3) * 8;
            *(uint4*)&As[row][c4] =
                *(const uint4*)&A[(size_t)(m0 + row) * K + k0 + c4];
        }
#endif
        // ---- stage B tile 32x128, transposed into [n][k] ----
#pragma unroll
        for (int i = 0; i < 2; ++i) {
            int idx = tid * 2 + i;            // 0..511
            int kk = idx >> 4, nn = (idx & 15) * 8;
            v8h val = *(const v8h*)&Bw[(size_t)(k0 + kk) * N + n0 + nn];
#pragma unroll
            for (int j = 0; j < 8; ++j) Bs[nn + j][kk] = val[j];
        }
        // prefetch next K-slab into caches while we compute
        if (k0 + 32 < K) {
            __builtin_prefetch(&Bw[(size_t)(k0 + 32 + (tid & 31)) * N + n0], 0, 0);
        }
        __syncthreads();

        v16h a[4], bf[2];
#pragma unroll
        for (int fm = 0; fm < 4; ++fm)
            a[fm] = frag_from_lds(&As[wm * 64 + fm * 16 + r16][kb]);
#pragma unroll
        for (int fn = 0; fn < 2; ++fn)
            bf[fn] = frag_from_lds(&Bs[wn * 32 + fn * 16 + r16][kb]);
#pragma unroll
        for (int fm = 0; fm < 4; ++fm)
#pragma unroll
            for (int fn = 0; fn < 2; ++fn)
                c[fm][fn] = __builtin_amdgcn_wmma_f32_16x16x32_f16(
                    false, a[fm], false, bf[fn], (short)0, c[fm][fn], false, false);
        __syncthreads();
    }

    // ---- epilogue: bias, activation, residual, store ----
#pragma unroll
    for (int fm = 0; fm < 4; ++fm)
#pragma unroll
        for (int fn = 0; fn < 2; ++fn) {
            int gmb = m0 + wm * 64 + fm * 16 + (lane >> 4) * 8;
            int gn  = n0 + wn * 32 + fn * 16 + r16;
            float bb = bias[gn];
#pragma unroll
            for (int r = 0; r < 8; ++r) {
                int gm = gmb + r;
                float v = c[fm][fn][r] + bb;
                if (ACT == 1) v = 0.5f * v * (1.0f + erff(v * 0.70710678118f));
                if (ADDRES) v += res[(size_t)gm * N + gn];
                if (OUTF32) ((float*)Cout)[(size_t)gm * N + gn] = v;
                else ((_Float16*)Cout)[(size_t)gm * N + gn] = (_Float16)v;
            }
        }
}

// ---------------------------------------------------------------------------
// Fused attention with cross-HEAD softmax (the reference's axis=1 quirk).
// Block = 128 threads (4 waves), one (batch, 16-query tile) per block.
//   per 32-k step:
//     stage K/V tiles (all 12 heads) -> LDS
//     waves 0..3 compute score tiles for heads 3w..3w+2 via WMMA -> Sc (f32)
//     elementwise 12-way softmax over heads -> Wt (f16)
//     each wave owns d-chunk wave*16: acc[h] += Wt[h] x V[h] (WMMA)
// Dynamic LDS layout (bytes):
//   Kt  [12][32][64] f16   @ 0       (49152)
//   Vt  [12][64][32] f16   @ 49152   (49152)   (V transposed: [d][k])
//   Sc  [12][16][32] f32   @ 98304   (24576)
//   Wt  [12][16][32] f16   @ 122880  (12288)
// ---------------------------------------------------------------------------
#define ATTN_LDS_BYTES 135168

__global__ void __launch_bounds__(128)
attention_kernel(const _Float16* __restrict__ Qf, const _Float16* __restrict__ Kf,
                 const _Float16* __restrict__ Vf, _Float16* __restrict__ Out) {
    extern __shared__ char smem[];
    _Float16* Kt = (_Float16*)smem;                       // [h][k][d]
    _Float16* Vt = (_Float16*)(smem + 49152);             // [h][d][k]
    float*    Sc = (float*)   (smem + 98304);             // [h][q][k]
    _Float16* Wt = (_Float16*)(smem + 122880);            // [h][q][k]

    const int tid  = threadIdx.x;
    const int lane = tid & 31, wave = tid >> 5;
    const int qt = blockIdx.x, bb = blockIdx.y;
    const int r16 = lane & 15, kb8 = (lane >> 4) * 8;
    const float scale = 0.03608439182f;                   // 1/sqrt(768), per reference

    // Q fragments for this wave's 3 score-heads, held across the k loop.
    const size_t rowQ = (size_t)(bb * SEQ + qt * 16 + r16) * EMBED;
    v16h qa[3][2];
#pragma unroll
    for (int hh = 0; hh < 3; ++hh) {
        int h = wave * 3 + hh;
        const _Float16* p = Qf + rowQ + h * DEPTH;
        qa[hh][0] = cat8(*(const v8h*)(p + kb8),      *(const v8h*)(p + kb8 + 16));
        qa[hh][1] = cat8(*(const v8h*)(p + 32 + kb8), *(const v8h*)(p + 32 + kb8 + 16));
    }

    v8f acc[HEADS];
    {
        v8f z = {};
#pragma unroll
        for (int h = 0; h < HEADS; ++h) acc[h] = z;
    }

    for (int kt = 0; kt < SEQ / 32; ++kt) {
        // ---- stage K (row-major [k][d]) and V (transposed [d][k]) ----
#pragma unroll 4
        for (int i = 0; i < 24; ++i) {
            int gid = tid + i * 128;              // 0..3071 : (h, kpos, dgroup)
            int h  = gid >> 8;
            int rm = gid & 255;
            int kp = rm >> 3;
            int g  = (rm & 7) * 8;
            size_t src = (size_t)(bb * SEQ + kt * 32 + kp) * EMBED + h * DEPTH + g;
            *(uint4*)&Kt[h * 2048 + kp * 64 + g] = *(const uint4*)&Kf[src];
            v8h vv = *(const v8h*)&Vf[src];
#pragma unroll
            for (int j = 0; j < 8; ++j) Vt[h * 2048 + (g + j) * 32 + kp] = vv[j];
        }
        __syncthreads();

        // ---- scores: this wave's 3 heads, 16q x 32k tile each ----
#pragma unroll
        for (int hh = 0; hh < 3; ++hh) {
            int h = wave * 3 + hh;
#pragma unroll
            for (int nt = 0; nt < 2; ++nt) {
                // preload both depth-chunk B fragments, then issue both WMMAs
                v16h bf0 = frag_from_lds(&Kt[h * 2048 + (nt * 16 + r16) * 64 + kb8]);
                v16h bf1 = frag_from_lds(&Kt[h * 2048 + (nt * 16 + r16) * 64 + 32 + kb8]);
                v8f s = {};
                s = __builtin_amdgcn_wmma_f32_16x16x32_f16(
                    false, qa[hh][0], false, bf0, (short)0, s, false, false);
                s = __builtin_amdgcn_wmma_f32_16x16x32_f16(
                    false, qa[hh][1], false, bf1, (short)0, s, false, false);
#pragma unroll
                for (int r = 0; r < 8; ++r) {
                    int q = r + (lane >> 4) * 8;
                    Sc[h * 512 + q * 32 + nt * 16 + r16] = s[r] * scale;
                }
            }
        }
        __syncthreads();

        // ---- softmax across the 12 heads, elementwise per (q,k) ----
#pragma unroll
        for (int i = 0; i < 4; ++i) {
            int pos = tid + i * 128;              // 0..511 = q*32 + k
            float mx = -1e30f;
#pragma unroll
            for (int h = 0; h < HEADS; ++h) mx = fmaxf(mx, Sc[h * 512 + pos]);
            float e[HEADS], sum = 0.0f;
#pragma unroll
            for (int h = 0; h < HEADS; ++h) {
                e[h] = __expf(Sc[h * 512 + pos] - mx);
                sum += e[h];
            }
            float inv = 1.0f / sum;
#pragma unroll
            for (int h = 0; h < HEADS; ++h)
                Wt[h * 512 + pos] = (_Float16)(e[h] * inv);
        }
        __syncthreads();

        // ---- acc[h] += W[h](16q x 32k) x V[h](32k x 16d), d-chunk = wave*16 ----
#pragma unroll
        for (int h = 0; h < HEADS; ++h) {
            v16h wa = frag_from_lds(&Wt[h * 512 + r16 * 32 + kb8]);
            v16h vb = frag_from_lds(&Vt[h * 2048 + (wave * 16 + r16) * 32 + kb8]);
            acc[h] = __builtin_amdgcn_wmma_f32_16x16x32_f16(
                false, wa, false, vb, (short)0, acc[h], false, false);
        }
        __syncthreads();
    }

    // ---- store: heads re-concatenated, [tok][h*64 + d] ----
#pragma unroll
    for (int h = 0; h < HEADS; ++h)
#pragma unroll
        for (int r = 0; r < 8; ++r) {
            int q = qt * 16 + r + (lane >> 4) * 8;
            Out[(size_t)(bb * SEQ + q) * EMBED + h * DEPTH + wave * 16 + r16] =
                (_Float16)acc[h][r];
        }
}

// ---------------------------------------------------------------------------
// Host-side orchestration
// ---------------------------------------------------------------------------
extern "C" void kernel_launch(void* const* d_in, const int* in_sizes, int n_in,
                              void* d_out, int out_size, void* d_ws, size_t ws_size,
                              hipStream_t stream) {
    const float* x    = (const float*)d_in[0];
    const float* ln1g = (const float*)d_in[1];
    const float* ln1b = (const float*)d_in[2];
    const float* wq   = (const float*)d_in[3];
    const float* bq   = (const float*)d_in[4];
    const float* wk   = (const float*)d_in[5];
    const float* bk   = (const float*)d_in[6];
    const float* wv   = (const float*)d_in[7];
    const float* bv   = (const float*)d_in[8];
    const float* wo   = (const float*)d_in[9];
    const float* bo   = (const float*)d_in[10];
    const float* ln2g = (const float*)d_in[11];
    const float* ln2b = (const float*)d_in[12];
    const float* w1   = (const float*)d_in[13];
    const float* b1   = (const float*)d_in[14];
    const float* w2   = (const float*)d_in[15];
    const float* b2   = (const float*)d_in[16];

    char* ws = (char*)d_ws;
    size_t off = 0;
    auto carve = [&](size_t bytes) -> void* {
        void* p = ws + off;
        off += (bytes + 255) & ~(size_t)255;
        return p;
    };
    const size_t actH = (size_t)NTOK * EMBED * sizeof(_Float16);        // 12.6 MB
    _Float16* hf   = (_Float16*)carve(actH);
    _Float16* qf   = (_Float16*)carve(actH);
    _Float16* kf   = (_Float16*)carve(actH);
    _Float16* vf   = (_Float16*)carve(actH);
    _Float16* attn = (_Float16*)carve(actH);
    _Float16* h2f  = (_Float16*)carve(actH);
    float*    msa  = (float*)   carve((size_t)NTOK * EMBED * sizeof(float));
    _Float16* m1f  = (_Float16*)carve((size_t)NTOK * HIDDEN * sizeof(_Float16));
    _Float16* wqh  = (_Float16*)carve((size_t)EMBED * EMBED * sizeof(_Float16));
    _Float16* wkh  = (_Float16*)carve((size_t)EMBED * EMBED * sizeof(_Float16));
    _Float16* wvh  = (_Float16*)carve((size_t)EMBED * EMBED * sizeof(_Float16));
    _Float16* woh  = (_Float16*)carve((size_t)EMBED * EMBED * sizeof(_Float16));
    _Float16* w1h  = (_Float16*)carve((size_t)EMBED * HIDDEN * sizeof(_Float16));
    _Float16* w2h  = (_Float16*)carve((size_t)HIDDEN * EMBED * sizeof(_Float16));

    // --- weight conversion f32 -> f16 ---
    {
        int nE = EMBED * EMBED, nH = EMBED * HIDDEN;
        f32_to_f16_kernel<<<(nE + 255) / 256, 256, 0, stream>>>(wq, wqh, nE);
        f32_to_f16_kernel<<<(nE + 255) / 256, 256, 0, stream>>>(wk, wkh, nE);
        f32_to_f16_kernel<<<(nE + 255) / 256, 256, 0, stream>>>(wv, wvh, nE);
        f32_to_f16_kernel<<<(nE + 255) / 256, 256, 0, stream>>>(wo, woh, nE);
        f32_to_f16_kernel<<<(nH + 255) / 256, 256, 0, stream>>>(w1, w1h, nH);
        f32_to_f16_kernel<<<(nH + 255) / 256, 256, 0, stream>>>(w2, w2h, nH);
    }

    // --- LN1 ---
    ln_kernel<<<NTOK, 256, 0, stream>>>(x, nullptr, ln1g, ln1b, hf);

    // --- QKV projections ---
    dim3 gP(EMBED / 128, NTOK / 128);          // (6, 64)
    gemm_wmma<0, 0, 0><<<gP, 256, 0, stream>>>(hf, wqh, bq, nullptr, qf,
                                               NTOK, EMBED, EMBED);
    gemm_wmma<0, 0, 0><<<gP, 256, 0, stream>>>(hf, wkh, bk, nullptr, kf,
                                               NTOK, EMBED, EMBED);
    gemm_wmma<0, 0, 0><<<gP, 256, 0, stream>>>(hf, wvh, bv, nullptr, vf,
                                               NTOK, EMBED, EMBED);

    // --- fused attention (cross-head softmax) ---
    (void)hipFuncSetAttribute((const void*)attention_kernel,
                              hipFuncAttributeMaxDynamicSharedMemorySize,
                              ATTN_LDS_BYTES);
    dim3 gA(SEQ / 16, BATCH);                  // (128, 4)
    attention_kernel<<<gA, 128, ATTN_LDS_BYTES, stream>>>(qf, kf, vf, attn);

    // --- output projection -> msa (f32) ---
    gemm_wmma<0, 1, 0><<<gP, 256, 0, stream>>>(attn, woh, bo, nullptr, msa,
                                               NTOK, EMBED, EMBED);

    // --- LN2 of (msa + x) ---
    ln_kernel<<<NTOK, 256, 0, stream>>>(msa, x, ln2g, ln2b, h2f);

    // --- MLP: GEMM+GELU then GEMM + bias + msa residual -> d_out ---
    dim3 g1(HIDDEN / 128, NTOK / 128);         // (24, 64)
    gemm_wmma<1, 0, 0><<<g1, 256, 0, stream>>>(h2f, w1h, b1, nullptr, m1f,
                                               NTOK, HIDDEN, EMBED);
    gemm_wmma<0, 1, 1><<<gP, 256, 0, stream>>>(m1f, w2h, b2, msa, (float*)d_out,
                                               NTOK, EMBED, HIDDEN);
}
